// SlotAttention_61521111548149
// MI455X (gfx1250) — matrix operand order repaired and checked by reading, not compiled, not run on hardware
//
#include <hip/hip_runtime.h>
#include <hip/hip_bf16.h>

// ---------------------------------------------------------------------------
// Slot Attention for MI455X (gfx1250): bf16 WMMA everywhere, f32 accumulate.
// B=32, N=4096, K=8 slots, HID=D=256, DM=512, ITERS=3, SCALE=1/16, TEMP=1.
// ---------------------------------------------------------------------------

typedef __bf16 bf16_t;
typedef __attribute__((ext_vector_type(16))) __bf16 v16bf;
typedef __attribute__((ext_vector_type(8)))  __bf16 v8bf;
typedef __attribute__((ext_vector_type(8)))  float  v8f;

#define WMMA_BF16(A, Bm, C) \
  __builtin_amdgcn_wmma_f32_16x16x32_bf16(false, (A), false, (Bm), (short)0, (C), false, false)

__device__ __forceinline__ v16bf cat8(v8bf lo, v8bf hi) {
  return __builtin_shufflevector(lo, hi, 0,1,2,3,4,5,6,7,8,9,10,11,12,13,14,15);
}

// A-matrix (16x32 bf16) per-lane load from a row-major row pointer.
// A layout: lane<16 -> K=ks*32+0..15, lane>=16 -> +16 (matches B).
__device__ __forceinline__ v16bf ld_a(const bf16_t* row, int hi, int ks) {
  const int c = ks * 32 + (hi ? 16 : 0);
  v8bf lo = *(const v8bf*)(row + c);
  v8bf hh = *(const v8bf*)(row + c + 8);
  return cat8(lo, hh);
}

// B-matrix (32x16 bf16) per-lane load: lane<16 -> K=ks*32+0..15, lane>=16 -> +16.
// "row" points at the column's contiguous K-major data (row-major k / transposed v).
__device__ __forceinline__ v16bf ld_b(const bf16_t* row, int hi, int ks) {
  const int c = ks * 32 + (hi ? 16 : 0);
  v8bf lo = *(const v8bf*)(row + c);
  v8bf hh = *(const v8bf*)(row + c + 8);
  return cat8(lo, hh);
}

// Pre-packed weight B-panel: one aligned 32-byte load per lane.
__device__ __forceinline__ v16bf ld_panel(const bf16_t* Wp, int nt, int nks, int ks, int lane) {
  return *(const v16bf*)(Wp + ((size_t)(nt * nks + ks) * 32 + lane) * 16);
}

__device__ __forceinline__ float wave_sum(float v) {
  #pragma unroll
  for (int off = 16; off >= 1; off >>= 1) v += __shfl_xor(v, off, 32);
  return v;
}

__device__ __forceinline__ v8f vzero() {
  v8f z = {0.f, 0.f, 0.f, 0.f, 0.f, 0.f, 0.f, 0.f};
  return z;
}

// ---------------------------------------------------------------------------
// Pack f32 weight W[KD][ND] (row-major, [in,out]) into bf16 B-panels.
// panel = (nt * (KD/32) + ks); element (lane, h):
//   kk = ks*32 + (lane>=16 ? 16 : 0) + h,  nn = nt*16 + (lane&15)
// ---------------------------------------------------------------------------
__global__ void pack_kernel(const float* __restrict__ W, bf16_t* __restrict__ out,
                            int KD, int ND) {
  const int total = KD * ND;
  const int nks = KD >> 5;
  for (int i = blockIdx.x * blockDim.x + threadIdx.x; i < total;
       i += gridDim.x * blockDim.x) {
    const int h = i & 15;
    const int lane = (i >> 4) & 31;
    const int panel = i >> 9;
    const int ks = panel % nks;
    const int nt = panel / nks;
    const int kk = ks * 32 + ((lane >= 16) ? 16 : 0) + h;
    const int nn = nt * 16 + (lane & 15);
    out[i] = (bf16_t)W[(size_t)kk * ND + nn];
  }
}

// slots = mu + sigma * noise   [32*8*256]
__global__ void init_slots_kernel(const float* __restrict__ mu, const float* __restrict__ sigma,
                                  const float* __restrict__ noise, float* __restrict__ slots) {
  const int i = blockIdx.x * blockDim.x + threadIdx.x;
  if (i < 32 * 8 * 256) {
    const int d = i & 255;
    slots[i] = mu[d] + sigma[d] * noise[i];
  }
}

// ---------------------------------------------------------------------------
// Fused: LayerNorm(inputs) -> bf16 tiles in LDS -> k = x@Wk+bk (row-major bf16),
//                                                  v = x@Wv+bv (transposed bf16)
// 64 rows (4 M-tiles) per block for 4x higher weight reuse:
//   grid = B*N/64 = 2048 blocks, 256 threads (8 waves).
//   Waves 0..3: k d-tiles (w*4..w*4+3); waves 4..7: v d-tiles ((w-4)*4..+3).
//   Per K-step each wave keeps its 4 B panels in registers and sweeps 4 A tiles.
// ---------------------------------------------------------------------------
__global__ __launch_bounds__(256) void kv_kernel(
    const float* __restrict__ x, const float* __restrict__ lng, const float* __restrict__ lnb,
    const bf16_t* __restrict__ Wk_p, const bf16_t* __restrict__ Wv_p,
    const float* __restrict__ bk, const float* __restrict__ bv,
    bf16_t* __restrict__ kb, bf16_t* __restrict__ vt) {
  __shared__ bf16_t xlds[64][256];   // 32 KB
  __shared__ bf16_t klds[64][256];   // 32 KB
  const int blk = blockIdx.x;        // 2048
  const int b = blk >> 6;
  const int nbase = (blk & 63) * 64;
  const size_t rowbase = (size_t)blk * 64;
  const int tid = threadIdx.x, wave = tid >> 5, lane = tid & 31;
  const int m = lane & 15, hi = lane >> 4;

  // LayerNorm 64 rows (wave handles 8 rows; 8 floats per lane per row)
  for (int rr = 0; rr < 8; ++rr) {
    const int row = wave * 8 + rr;
    const float* xr = x + (rowbase + row) * 256;
    float xv[8], s = 0.f, s2 = 0.f;
    #pragma unroll
    for (int e = 0; e < 8; ++e) { float v = xr[lane * 8 + e]; xv[e] = v; s += v; s2 += v * v; }
    s = wave_sum(s); s2 = wave_sum(s2);
    const float mean = s * (1.f / 256.f);
    const float var = s2 * (1.f / 256.f) - mean * mean;
    const float rstd = rsqrtf(var + 1e-5f);
    #pragma unroll
    for (int e = 0; e < 8; ++e) {
      const int cc = lane * 8 + e;
      xlds[row][cc] = (bf16_t)((xv[e] - mean) * rstd * lng[cc] + lnb[cc]);
    }
  }
  __syncthreads();

  const int isv = wave >> 2;                       // 0: k-waves, 1: v-waves
  const bf16_t* Wp = isv ? Wv_p : Wk_p;
  const int dt0 = (wave & 3) * 4;

  v8f acc[4][4];                                   // [row-tile][d-tile]
  #pragma unroll
  for (int rt = 0; rt < 4; ++rt)
    #pragma unroll
    for (int t = 0; t < 4; ++t) acc[rt][t] = vzero();

  for (int ks = 0; ks < 8; ++ks) {
    v16bf bp[4];
    #pragma unroll
    for (int t = 0; t < 4; ++t) bp[t] = ld_panel(Wp, dt0 + t, 8, ks, lane);
    #pragma unroll
    for (int rt = 0; rt < 4; ++rt) {
      v16bf a = ld_a(&xlds[rt * 16 + m][0], hi, ks);
      #pragma unroll
      for (int t = 0; t < 4; ++t)
        acc[rt][t] = WMMA_BF16(a, bp[t], acc[rt][t]);
    }
  }

  if (!isv) {
    // k epilogue: bias, bf16, transpose through LDS for coalesced row-major stores
    #pragma unroll
    for (int t = 0; t < 4; ++t) {
      const int d = (dt0 + t) * 16 + m;
      const float bias = bk[d];
      #pragma unroll
      for (int rt = 0; rt < 4; ++rt)
        #pragma unroll
        for (int r = 0; r < 8; ++r)
          klds[rt * 16 + r + hi * 8][d] = (bf16_t)(acc[rt][t][r] + bias);
    }
  } else {
    // v epilogue: transposed store, one packed 16B store per lane per tile
    #pragma unroll
    for (int t = 0; t < 4; ++t) {
      const int d = (dt0 + t) * 16 + m;
      const float bias = bv[d];
      #pragma unroll
      for (int rt = 0; rt < 4; ++rt) {
        v8bf pk;
        #pragma unroll
        for (int r = 0; r < 8; ++r) pk[r] = (bf16_t)(acc[rt][t][r] + bias);
        *(v8bf*)(vt + ((size_t)b * 256 + d) * 4096 + nbase + rt * 16 + hi * 8) = pk;
      }
    }
  }
  __syncthreads();
  // cooperative row-major k store: 64 rows x 32 chunks of 8 halfs
  for (int idx = tid; idx < 2048; idx += 256) {
    const int row = idx >> 5, ch = idx & 31;
    *(uint4*)(kb + (rowbase + row) * 256 + ch * 8) = *(const uint4*)(&klds[row][ch * 8]);
  }
}

// ---------------------------------------------------------------------------
// q = LayerNorm(slots)@Wq + bq  (f32 VALU; tiny), also store slots_prev bf16
// and zero attn_sum. grid = 32 (one per batch), 256 threads (8 waves = 8 slots).
// ---------------------------------------------------------------------------
__global__ __launch_bounds__(256) void qproj_kernel(
    const float* __restrict__ slots, const float* __restrict__ lng, const float* __restrict__ lnb,
    const float* __restrict__ Wq, const float* __restrict__ bq,
    bf16_t* __restrict__ qb, bf16_t* __restrict__ hb, float* __restrict__ attn_sum) {
  __shared__ float sl[8][256];
  const int b = blockIdx.x, tid = threadIdx.x, wave = tid >> 5, lane = tid & 31;
  const float* row = slots + ((size_t)b * 8 + wave) * 256;
  float xv[8], s = 0.f, s2 = 0.f;
  #pragma unroll
  for (int e = 0; e < 8; ++e) { float v = row[lane * 8 + e]; xv[e] = v; s += v; s2 += v * v; }
  s = wave_sum(s); s2 = wave_sum(s2);
  const float mean = s * (1.f / 256.f);
  const float rstd = rsqrtf(s2 * (1.f / 256.f) - mean * mean + 1e-5f);
  #pragma unroll
  for (int e = 0; e < 8; ++e) {
    const int cc = lane * 8 + e;
    hb[((size_t)b * 8 + wave) * 256 + cc] = (bf16_t)xv[e];                 // raw slots_prev (GRU h)
    sl[wave][cc] = (xv[e] - mean) * rstd * lng[cc] + lnb[cc];
  }
  if (tid < 8) attn_sum[b * 8 + tid] = 0.f;
  __syncthreads();

  float acc[8] = {0.f, 0.f, 0.f, 0.f, 0.f, 0.f, 0.f, 0.f};
  for (int d = 0; d < 256; ++d) {
    const float sv = sl[wave][d];
    const float* wr = Wq + (size_t)d * 256;
    #pragma unroll
    for (int j = 0; j < 8; ++j) acc[j] += sv * wr[lane + 32 * j];
  }
  #pragma unroll
  for (int j = 0; j < 8; ++j) {
    const int dout = lane + 32 * j;
    qb[((size_t)b * 16 + wave) * 256 + dout] = (bf16_t)(acc[j] + bq[dout]);
  }
}

// ---------------------------------------------------------------------------
// dots = q@k^T * SCALE ; inverted softmax over 8 slots (in-lane!) ; +EPS ;
// accumulate per-slot sums. grid = 32*256 = 8192 blocks of one wave.
// ---------------------------------------------------------------------------
__global__ __launch_bounds__(32) void dots_kernel(
    const bf16_t* __restrict__ qb, const bf16_t* __restrict__ kb,
    bf16_t* __restrict__ attn, float* __restrict__ attn_sum) {
  const int blk = blockIdx.x, b = blk >> 8, nt = blk & 255;
  const int lane = threadIdx.x, m = lane & 15, hi = lane >> 4;
  const bf16_t* ar = qb + ((size_t)b * 16 + m) * 256;                 // A: q rows (slots)
  const bf16_t* br = kb + ((size_t)b * 4096 + nt * 16 + m) * 256;     // B: k rows (keys)
  v8f acc = vzero();
  for (int ks = 0; ks < 8; ++ks)
    acc = WMMA_BF16(ld_a(ar, hi, ks), ld_b(br, hi, ks), acc);

  // lanes 0..15 hold all 8 slot logits for key n = nt*16+lane
  float av[8], mx = -1e30f;
  #pragma unroll
  for (int r = 0; r < 8; ++r) { float v = acc[r] * 0.0625f; av[r] = v; mx = fmaxf(mx, v); }
  float sum = 0.f;
  #pragma unroll
  for (int r = 0; r < 8; ++r) { float e = __expf(av[r] - mx); av[r] = e; sum += e; }
  const float inv = 1.f / sum;
  #pragma unroll
  for (int r = 0; r < 8; ++r) av[r] = av[r] * inv + 1e-8f;

  #pragma unroll
  for (int r = 0; r < 8; ++r) {
    float part = hi ? 0.f : av[r];
    part = wave_sum(part);
    if (lane == 0) atomicAdd(&attn_sum[b * 8 + r], part);
  }
  if (!hi) {
    #pragma unroll
    for (int r = 0; r < 8; ++r)
      attn[((size_t)b * 16 + r) * 4096 + nt * 16 + m] = (bf16_t)av[r];
  }
}

// ---------------------------------------------------------------------------
// updates = (attn @ v) / attn_sum  -> bf16 [256 rows][256]
// grid = 32*16 = 512 blocks of one wave; 128-step WMMA chain over n=4096.
// ---------------------------------------------------------------------------
__global__ __launch_bounds__(32) void upd_kernel(
    const bf16_t* __restrict__ attn, const bf16_t* __restrict__ vt,
    const float* __restrict__ attn_sum, bf16_t* __restrict__ ub) {
  const int blk = blockIdx.x, b = blk >> 4, dt = blk & 15;
  const int lane = threadIdx.x, m = lane & 15, hi = lane >> 4;
  const bf16_t* ar = attn + ((size_t)b * 16 + m) * 4096;              // A rows = slots
  const bf16_t* br = vt + ((size_t)b * 256 + dt * 16 + m) * 4096;     // B cols = d (K-major)
  v8f acc = vzero();
  for (int ks = 0; ks < 128; ++ks)
    acc = WMMA_BF16(ld_a(ar, hi, ks), ld_b(br, hi, ks), acc);
  if (!hi) {
    const int d = dt * 16 + m;
    #pragma unroll
    for (int r = 0; r < 8; ++r) {
      const float inv = 1.f / attn_sum[b * 8 + r];
      ub[((size_t)b * 8 + r) * 256 + d] = (bf16_t)(acc[r] * inv);
    }
  }
}

// ---------------------------------------------------------------------------
// GRU: gi = u@W_ih, gh = h@W_hh (both WMMA), gate math, slots update, then
// LayerNorm(h_new) -> mlpin bf16. grid = 16 row-tiles, 256 threads (8 waves x 6 tiles).
// LDS: grz[16][512] (i+h for r,z) | gin[16][256] | ghn[16][256] = 64 KB.
// ---------------------------------------------------------------------------
__global__ __launch_bounds__(256) void gru_kernel(
    const bf16_t* __restrict__ ub, const bf16_t* __restrict__ hb,
    const bf16_t* __restrict__ Wih_p, const bf16_t* __restrict__ Whh_p,
    const float* __restrict__ b_ih, const float* __restrict__ b_hh,
    const float* __restrict__ lng, const float* __restrict__ lnb,
    float* __restrict__ slots, bf16_t* __restrict__ mlpin) {
  __shared__ float smem[16384];
  float* grz = smem;                       // [16][512]
  float* gin = smem + 16 * 512;            // [16][256]
  float* ghn = smem + 16 * 512 + 16 * 256; // [16][256], reused as h_new
  const int rt = blockIdx.x;
  const int tid = threadIdx.x, wave = tid >> 5, lane = tid & 31;
  const int m = lane & 15, hi = lane >> 4;
  const bf16_t* au = ub + (size_t)(rt * 16 + m) * 256;
  const bf16_t* ah = hb + (size_t)(rt * 16 + m) * 256;

  v8f ai[6], ahc[6];
  #pragma unroll
  for (int t = 0; t < 6; ++t) { ai[t] = vzero(); ahc[t] = vzero(); }
  for (int ks = 0; ks < 8; ++ks) {
    v16bf Au = ld_a(au, hi, ks);
    v16bf Ah = ld_a(ah, hi, ks);
    #pragma unroll
    for (int t = 0; t < 6; ++t) {
      const int nt = wave * 6 + t;
      ai[t]  = WMMA_BF16(Au, ld_panel(Wih_p, nt, 8, ks, lane), ai[t]);
      ahc[t] = WMMA_BF16(Ah, ld_panel(Whh_p, nt, 8, ks, lane), ahc[t]);
    }
  }
  #pragma unroll
  for (int t = 0; t < 6; ++t) {
    const int col = (wave * 6 + t) * 16 + m;
    if (col < 512) {
      #pragma unroll
      for (int r = 0; r < 8; ++r) grz[(r + hi * 8) * 512 + col] = ai[t][r] + ahc[t][r];
    } else {
      const int cj = col - 512;
      #pragma unroll
      for (int r = 0; r < 8; ++r) {
        gin[(r + hi * 8) * 256 + cj] = ai[t][r];
        ghn[(r + hi * 8) * 256 + cj] = ahc[t][r];
      }
    }
  }
  __syncthreads();

  for (int idx = tid; idx < 4096; idx += 256) {
    const int mm = idx >> 8, j = idx & 255;
    const size_t grow = (size_t)rt * 16 + mm;
    float r = grz[mm * 512 + j] + b_ih[j] + b_hh[j];
    r = 1.f / (1.f + __expf(-r));
    float z = grz[mm * 512 + 256 + j] + b_ih[256 + j] + b_hh[256 + j];
    z = 1.f / (1.f + __expf(-z));
    const float n = tanhf(gin[mm * 256 + j] + b_ih[512 + j] + r * (ghn[mm * 256 + j] + b_hh[512 + j]));
    const float hp = slots[grow * 256 + j];
    const float h = (1.f - z) * n + z * hp;
    slots[grow * 256 + j] = h;
    ghn[mm * 256 + j] = h;   // reuse as h_new
  }
  __syncthreads();

  // LayerNorm(h_new) -> mlpin bf16 (wave handles 2 rows)
  #pragma unroll
  for (int rr = 0; rr < 2; ++rr) {
    const int mm = wave * 2 + rr;
    float hv[8], s = 0.f, s2 = 0.f;
    #pragma unroll
    for (int e = 0; e < 8; ++e) { float v = ghn[mm * 256 + lane * 8 + e]; hv[e] = v; s += v; s2 += v * v; }
    s = wave_sum(s); s2 = wave_sum(s2);
    const float mean = s * (1.f / 256.f);
    const float rstd = rsqrtf(s2 * (1.f / 256.f) - mean * mean + 1e-5f);
    #pragma unroll
    for (int e = 0; e < 8; ++e) {
      const int cc = lane * 8 + e;
      mlpin[(size_t)(rt * 16 + mm) * 256 + cc] = (bf16_t)((hv[e] - mean) * rstd * lng[cc] + lnb[cc]);
    }
  }
}

// hidden = relu(mlpin@W1 + b1) -> bf16 [256][512]; 16 blocks, 8 waves x 4 tiles
__global__ __launch_bounds__(256) void mlp1_kernel(
    const bf16_t* __restrict__ mlpin, const bf16_t* __restrict__ W1_p,
    const float* __restrict__ b1, bf16_t* __restrict__ hid) {
  const int rt = blockIdx.x, tid = threadIdx.x, wave = tid >> 5, lane = tid & 31;
  const int m = lane & 15, hi = lane >> 4;
  const bf16_t* ar = mlpin + (size_t)(rt * 16 + m) * 256;
  v8f acc[4];
  #pragma unroll
  for (int t = 0; t < 4; ++t) acc[t] = vzero();
  for (int ks = 0; ks < 8; ++ks) {
    v16bf a = ld_a(ar, hi, ks);
    #pragma unroll
    for (int t = 0; t < 4; ++t)
      acc[t] = WMMA_BF16(a, ld_panel(W1_p, wave * 4 + t, 8, ks, lane), acc[t]);
  }
  #pragma unroll
  for (int t = 0; t < 4; ++t) {
    const int col = (wave * 4 + t) * 16 + m;
    const float bias = b1[col];
    #pragma unroll
    for (int r = 0; r < 8; ++r) {
      float v = acc[t][r] + bias;
      v = v > 0.f ? v : 0.f;
      hid[(size_t)(rt * 16 + r + hi * 8) * 512 + col] = (bf16_t)v;
    }
  }
}

// slots += hidden@W2 + b2 ; 16 blocks, 8 waves x 2 tiles, K-chain 16
__global__ __launch_bounds__(256) void mlp2_kernel(
    const bf16_t* __restrict__ hid, const bf16_t* __restrict__ W2_p,
    const float* __restrict__ b2, float* __restrict__ slots) {
  const int rt = blockIdx.x, tid = threadIdx.x, wave = tid >> 5, lane = tid & 31;
  const int m = lane & 15, hi = lane >> 4;
  const bf16_t* ar = hid + (size_t)(rt * 16 + m) * 512;
  v8f acc[2];
  acc[0] = vzero(); acc[1] = vzero();
  for (int ks = 0; ks < 16; ++ks) {
    v16bf a = ld_a(ar, hi, ks);
    #pragma unroll
    for (int t = 0; t < 2; ++t)
      acc[t] = WMMA_BF16(a, ld_panel(W2_p, wave * 2 + t, 16, ks, lane), acc[t]);
  }
  #pragma unroll
  for (int t = 0; t < 2; ++t) {
    const int d = (wave * 2 + t) * 16 + m;
    const float bias = b2[d];
    #pragma unroll
    for (int r = 0; r < 8; ++r) {
      const size_t grow = (size_t)rt * 16 + r + hi * 8;
      slots[grow * 256 + d] += acc[t][r] + bias;
    }
  }
}

// ---------------------------------------------------------------------------
extern "C" void kernel_launch(void* const* d_in, const int* in_sizes, int n_in,
                              void* d_out, int out_size, void* d_ws, size_t ws_size,
                              hipStream_t stream) {
  const float* inputs   = (const float*)d_in[0];
  const float* noise    = (const float*)d_in[1];
  const float* mu       = (const float*)d_in[2];
  const float* sigma    = (const float*)d_in[3];
  const float* ln_in_g  = (const float*)d_in[4];
  const float* ln_in_b  = (const float*)d_in[5];
  const float* ln_sl_g  = (const float*)d_in[6];
  const float* ln_sl_b  = (const float*)d_in[7];
  const float* ln_ml_g  = (const float*)d_in[8];
  const float* ln_ml_b  = (const float*)d_in[9];
  const float* Wq       = (const float*)d_in[10];
  const float* bq       = (const float*)d_in[11];
  const float* Wk       = (const float*)d_in[12];
  const float* bk       = (const float*)d_in[13];
  const float* Wv       = (const float*)d_in[14];
  const float* bv       = (const float*)d_in[15];
  const float* W_ih     = (const float*)d_in[16];
  const float* W_hh     = (const float*)d_in[17];
  const float* b_ih     = (const float*)d_in[18];
  const float* b_hh     = (const float*)d_in[19];
  const float* W1       = (const float*)d_in[20];
  const float* b1       = (const float*)d_in[21];
  const float* W2       = (const float*)d_in[22];
  const float* b2       = (const float*)d_in[23];

  char* ws = (char*)d_ws;
  auto take = [&](size_t bytes) -> char* {
    char* p = ws;
    ws += (bytes + 255) & ~(size_t)255;
    return p;
  };
  bf16_t* kb      = (bf16_t*)take((size_t)32 * 4096 * 256 * 2);   // k row-major bf16
  bf16_t* vt      = (bf16_t*)take((size_t)32 * 256 * 4096 * 2);   // v transposed bf16
  bf16_t* qb      = (bf16_t*)take((size_t)32 * 16 * 256 * 2);     // q (rows 8..15 unused)
  bf16_t* attn    = (bf16_t*)take((size_t)32 * 16 * 4096 * 2);    // attn (rows 8..15 unused)
  float*  attnsum = (float*) take((size_t)32 * 8 * 4);
  float*  slots   = (float*) take((size_t)32 * 8 * 256 * 4);
  bf16_t* ub      = (bf16_t*)take((size_t)256 * 256 * 2);         // updates bf16
  bf16_t* hb      = (bf16_t*)take((size_t)256 * 256 * 2);         // slots_prev bf16
  bf16_t* mlpin   = (bf16_t*)take((size_t)256 * 256 * 2);
  bf16_t* hid     = (bf16_t*)take((size_t)256 * 512 * 2);
  bf16_t* Wk_p    = (bf16_t*)take((size_t)256 * 256 * 2);
  bf16_t* Wv_p    = (bf16_t*)take((size_t)256 * 256 * 2);
  bf16_t* Wih_p   = (bf16_t*)take((size_t)256 * 768 * 2);
  bf16_t* Whh_p   = (bf16_t*)take((size_t)256 * 768 * 2);
  bf16_t* W1_p    = (bf16_t*)take((size_t)256 * 512 * 2);
  bf16_t* W2_p    = (bf16_t*)take((size_t)512 * 256 * 2);

  // Pack weights into WMMA B-panels (recomputed every call; deterministic)
  pack_kernel<<<256, 256, 0, stream>>>(Wk,   Wk_p,  256, 256);
  pack_kernel<<<256, 256, 0, stream>>>(Wv,   Wv_p,  256, 256);
  pack_kernel<<<256, 256, 0, stream>>>(W_ih, Wih_p, 256, 768);
  pack_kernel<<<256, 256, 0, stream>>>(W_hh, Whh_p, 256, 768);
  pack_kernel<<<256, 256, 0, stream>>>(W1,   W1_p,  256, 512);
  pack_kernel<<<256, 256, 0, stream>>>(W2,   W2_p,  512, 256);

  init_slots_kernel<<<256, 256, 0, stream>>>(mu, sigma, noise, slots);

  // LN(inputs) + k/v projections (the heavy GEMM; 64 rows/block for weight reuse)
  kv_kernel<<<2048, 256, 0, stream>>>(inputs, ln_in_g, ln_in_b, Wk_p, Wv_p, bk, bv, kb, vt);

  for (int it = 0; it < 3; ++it) {
    qproj_kernel<<<32, 256, 0, stream>>>(slots, ln_sl_g, ln_sl_b, Wq, bq, qb, hb, attnsum);
    dots_kernel<<<8192, 32, 0, stream>>>(qb, kb, attn, attnsum);
    upd_kernel<<<512, 32, 0, stream>>>(attn, vt, attnsum, ub);
    gru_kernel<<<16, 256, 0, stream>>>(ub, hb, Wih_p, Whh_p, b_ih, b_hh,
                                       ln_ml_g, ln_ml_b, slots, mlpin);
    mlp1_kernel<<<16, 256, 0, stream>>>(mlpin, W1_p, b1, hid);
    mlp2_kernel<<<16, 256, 0, stream>>>(hid, W2_p, b2, slots);
  }

  hipMemcpyAsync(d_out, slots, (size_t)32 * 8 * 256 * 4, hipMemcpyDeviceToDevice, stream);
}
